// minLSTM_5712306503906
// MI455X (gfx1250) — matrix operand order repaired and checked by reading, not compiled, not run on hardware
//
#include <hip/hip_runtime.h>
#include <hip/hip_bf16.h>
#include <stdint.h>

// ---------------------------------------------------------------------------
// minLSTM forward for MI455X (gfx1250, wave32, WMMA + async-to-LDS).
// Pipeline: cvt(f32->bf16) -> WMMA GEMM1 (+gate epilogue) -> time scan -> WMMA GEMM2
// ---------------------------------------------------------------------------

typedef __attribute__((ext_vector_type(16))) __bf16 v16bf;
typedef __attribute__((ext_vector_type(8)))  float  v8f;
typedef __attribute__((ext_vector_type(4)))  int    v4i;

#define BM 128
#define BN 128
#define BK 32
#define LDS_STRIDE 40   // halves per tile row: 32 data + 8 pad (80B, 16B-aligned rows)

// ---- gfx1250 async global->LDS path (ASYNCcnt), guarded for portability ----
#if __has_builtin(__builtin_amdgcn_global_load_async_to_lds_b128)
#define HAVE_ASYNC_LDS 1
typedef __attribute__((address_space(1))) v4i gv4i_t;   // global int4
typedef __attribute__((address_space(3))) v4i lv4i_t;   // LDS int4
__device__ __forceinline__ gv4i_t* glob_cast(const void* p) {
    return (gv4i_t*)(uintptr_t)p;
}
__device__ __forceinline__ lv4i_t* lds_cast(void* p) {
    // flat shared pointers carry the LDS byte offset in addr[31:0] (ISA 10.2)
    return (lv4i_t*)(uint32_t)(uintptr_t)p;
}
#if __has_builtin(__builtin_amdgcn_s_wait_asynccnt)
#define WAIT_ASYNC() __builtin_amdgcn_s_wait_asynccnt(0)
#else
#define WAIT_ASYNC() asm volatile("s_wait_asynccnt 0x0" ::: "memory")
#endif
#else
#define HAVE_ASYNC_LDS 0
#define WAIT_ASYNC() do {} while (0)
#endif

__device__ __forceinline__ unsigned short f32_to_bf16(float f) {
    union { float f; uint32_t u; } c; c.f = f;
    uint32_t u = c.u;
    uint32_t r = u + 0x7FFFu + ((u >> 16) & 1u);      // round-to-nearest-even
    if ((u & 0x7F800000u) == 0x7F800000u) r = u;       // inf/nan: truncate
    return (unsigned short)(r >> 16);
}

__device__ __forceinline__ float logsig(float v) {     // -softplus(-v), stable
    return fminf(v, 0.0f) - log1pf(__expf(-fabsf(v)));
}
__device__ __forceinline__ float log_g_f(float v) {    // log_g from reference
    return (v >= 0.0f) ? __logf(v + 0.5f) : logsig(v);
}

// ---------------------------------------------------------------------------
// fp32 -> bf16 conversion, vectorized x4
// ---------------------------------------------------------------------------
__global__ void cvt_f32_bf16_x4(const float4* __restrict__ src,
                                ushort4* __restrict__ dst, int n4) {
    int i = blockIdx.x * blockDim.x + threadIdx.x;
    if (i < n4) {
        float4 f = src[i];
        ushort4 o;
        o.x = f32_to_bf16(f.x); o.y = f32_to_bf16(f.y);
        o.z = f32_to_bf16(f.z); o.w = f32_to_bf16(f.w);
        dst[i] = o;
    }
}

// ---------------------------------------------------------------------------
// Issue one 128x32 A tile + 128x32 B tile copy into LDS (per-thread: 4x b128).
// Async path: global_load_async_to_lds_b128 (no VGPR round-trip, ASYNCcnt).
// Fallback: batched register staging (all loads issued before stores).
// ---------------------------------------------------------------------------
__device__ __forceinline__ void tile_copy_issue(
    const unsigned short* __restrict__ A, const unsigned short* __restrict__ Bm,
    int K, int block_m, int block_n, int kt,
    unsigned short* As, unsigned short* Bs, int tid) {
#if HAVE_ASYNC_LDS
#pragma unroll
    for (int j = 0; j < 2; ++j) {
        int idx = tid * 2 + j;              // 0..511 chunks of 8 halves
        int row = idx >> 2;                 // 4 chunks per 32-half row
        int col = (idx & 3) * 8;
        __builtin_amdgcn_global_load_async_to_lds_b128(
            glob_cast(A + (size_t)(block_m + row) * K + kt + col),
            lds_cast(&As[row * LDS_STRIDE + col]), 0, 0);
        __builtin_amdgcn_global_load_async_to_lds_b128(
            glob_cast(Bm + (size_t)(block_n + row) * K + kt + col),
            lds_cast(&Bs[row * LDS_STRIDE + col]), 0, 0);
    }
#else
    uint4 ra[2], rb[2];
    int row[2], col[2];
#pragma unroll
    for (int j = 0; j < 2; ++j) {
        int idx = tid * 2 + j;
        row[j] = idx >> 2;
        col[j] = (idx & 3) * 8;
        ra[j] = *(const uint4*)(A  + (size_t)(block_m + row[j]) * K + kt + col[j]);
        rb[j] = *(const uint4*)(Bm + (size_t)(block_n + row[j]) * K + kt + col[j]);
    }
#pragma unroll
    for (int j = 0; j < 2; ++j) {
        *(uint4*)&As[row[j] * LDS_STRIDE + col[j]] = ra[j];
        *(uint4*)&Bs[row[j] * LDS_STRIDE + col[j]] = rb[j];
    }
#endif
}

// ---------------------------------------------------------------------------
// WMMA GEMM:  C[M,N] = A[M,K] * B[N,K]^T   (A,B bf16, acc fp32)
// MODE 0: store fp32 C to out0.
// MODE 1: val += bias[n]; split N into 3 planes of E=N/3:
//         logf = logsigmoid, logi = logsigmoid, logth = log_g.
// Block: 256 threads = 8 waves (2x4 wave grid); wave tile 64x32 = 4x2 WMMA tiles.
// Double-buffered LDS, async copy of tile k+1 overlapped with WMMAs of tile k.
// ---------------------------------------------------------------------------
template <int MODE>
__global__ __launch_bounds__(256)
void wmma_gemm(const unsigned short* __restrict__ A,
               const unsigned short* __restrict__ Bm,
               const float* __restrict__ bias,
               float* __restrict__ out0,
               float* __restrict__ out1,
               float* __restrict__ out2,
               int M, int N, int K) {
    __shared__ unsigned short As[2][BM * LDS_STRIDE];
    __shared__ unsigned short Bs[2][BN * LDS_STRIDE];

    const int tid   = threadIdx.x;
    const int lane  = tid & 31;
    const int wave  = tid >> 5;
    const int wm    = wave >> 2;            // 0..1  (64-row strip)
    const int wn    = wave & 3;             // 0..3  (32-col strip)
    const int block_m = blockIdx.y * BM;
    const int block_n = blockIdx.x * BN;

    const int lm    = lane & 15;            // row/col within 16x16 tile
    const int lk_hi = lane >> 4;            // K-half select, contiguous k = 16*lk_hi + s

    v8f acc[4][2];
#pragma unroll
    for (int mi = 0; mi < 4; ++mi)
#pragma unroll
        for (int ni = 0; ni < 2; ++ni)
            acc[mi][ni] = (v8f){0.f,0.f,0.f,0.f,0.f,0.f,0.f,0.f};

    union Frag { v16bf v; uint4 q[2]; };

    // prologue: bring in tile 0
    tile_copy_issue(A, Bm, K, block_m, block_n, 0, As[0], Bs[0], tid);
    WAIT_ASYNC();
    __syncthreads();

    int buf = 0;
    for (int kt = 0; kt < K; kt += BK, buf ^= 1) {
        // overlap: issue next tile's async copy into the other buffer
        if (kt + BK < K)
            tile_copy_issue(A, Bm, K, block_m, block_n, kt + BK,
                            As[buf ^ 1], Bs[buf ^ 1], tid);
        // warm GL2 two tiles ahead
        if (lane == 0 && kt + 2 * BK < K) {
            __builtin_prefetch(A  + (size_t)(block_m + wave * 16) * K + kt + 2 * BK, 0, 3);
            __builtin_prefetch(Bm + (size_t)(block_n + wave * 16) * K + kt + 2 * BK, 0, 3);
        }

        // --- B fragments: lane holds col n, 16 contiguous K halves (2x b128)
        Frag bfrag[2];
#pragma unroll
        for (int ni = 0; ni < 2; ++ni) {
            const unsigned short* p =
                &Bs[buf][(wn * 32 + ni * 16 + lm) * LDS_STRIDE + lk_hi * 16];
            bfrag[ni].q[0] = *(const uint4*)(p);
            bfrag[ni].q[1] = *(const uint4*)(p + 8);
        }
        // --- A fragments + WMMA
#pragma unroll
        for (int mi = 0; mi < 4; ++mi) {
            Frag afrag;
            const unsigned short* p =
                &As[buf][(wm * 64 + mi * 16 + lm) * LDS_STRIDE + lk_hi * 16];
            afrag.q[0] = *(const uint4*)(p);        // K = kh .. kh+7
            afrag.q[1] = *(const uint4*)(p + 8);    // K = kh+8 .. kh+15
#pragma unroll
            for (int ni = 0; ni < 2; ++ni) {
                acc[mi][ni] = __builtin_amdgcn_wmma_f32_16x16x32_bf16(
                    false, afrag.v, false, bfrag[ni].v,
                    (short)0, acc[mi][ni], false, false);
            }
        }

        WAIT_ASYNC();       // next tile's async copy has had the whole MMA phase
        __syncthreads();
    }

    // --- epilogue: C element (lane,vgpr) -> row m = hi8*8+v, col n = lm
    const int E = N / 3;   // only used in MODE 1
#pragma unroll
    for (int mi = 0; mi < 4; ++mi) {
#pragma unroll
        for (int ni = 0; ni < 2; ++ni) {
            const int n    = block_n + wn * 32 + ni * 16 + lm;
            const int mrow = block_m + wm * 64 + mi * 16 + lk_hi * 8;
#pragma unroll
            for (int v = 0; v < 8; ++v) {
                const int m  = mrow + v;
                float val    = acc[mi][ni][v];
                if (MODE == 0) {
                    out0[(size_t)m * N + n] = val;
                } else {
                    val += bias[n];
                    if (n < E)            out0[(size_t)m * E + n]           = logsig(val);
                    else if (n < 2 * E)   out1[(size_t)m * E + (n - E)]     = logsig(val);
                    else                  out2[(size_t)m * E + (n - 2 * E)] = log_g_f(val);
                }
            }
        }
    }
}

// ---------------------------------------------------------------------------
// Sequential log-space scan over T per (b,e) channel:
//   lh_t = logaddexp(lf_t + lh_{t-1}, li_t + lg_t),  lh_0 seed log(0.5)
//   h_t  = exp(lh_t)  -> bf16 for GEMM2
// 16384 channels, lanes map to consecutive e -> fully coalesced.
// ---------------------------------------------------------------------------
__global__ __launch_bounds__(256)
void scan_kernel(const float* __restrict__ logf, const float* __restrict__ logi,
                 const float* __restrict__ logth, unsigned short* __restrict__ hb,
                 int Bb, int T, int E) {
    int c = blockIdx.x * blockDim.x + threadIdx.x;
    if (c >= Bb * E) return;
    int b = c / E, e = c % E;
    float lh = -0.69314718056f;                     // log(0.5)
    size_t base = (size_t)b * T * E + e;
    for (int t = 0; t < T; ++t) {
        size_t idx = base + (size_t)t * E;
        float a  = logf[idx] + lh;
        float lv = logi[idx] + logth[idx];
        float mx = fmaxf(a, lv);
        lh = mx + log1pf(__expf(-fabsf(a - lv)));
        hb[idx] = f32_to_bf16(__expf(lh));
    }
}

// ---------------------------------------------------------------------------
extern "C" void kernel_launch(void* const* d_in, const int* in_sizes, int n_in,
                              void* d_out, int out_size, void* d_ws, size_t ws_size,
                              hipStream_t stream) {
    (void)in_sizes; (void)n_in; (void)out_size; (void)ws_size;

    const float* x  = (const float*)d_in[0];   // [B,T,D]
    const float* W  = (const float*)d_in[1];   // [3E,D]
    const float* b  = (const float*)d_in[2];   // [3E]
    const float* Wd = (const float*)d_in[3];   // [D,E]
    float* out = (float*)d_out;                // [B,T,D]

    const int Bb = 8, T = 2048, D = 1024, E = 2048;
    const int M = Bb * T;                      // 16384

    // workspace carve-up (~496 MB)
    char* ws = (char*)d_ws;
    unsigned short* xb  = (unsigned short*)ws;                 // [M,D]  bf16
    unsigned short* Wb  = xb  + (size_t)M * D;                 // [3E,D] bf16
    unsigned short* Wdb = Wb  + (size_t)3 * E * D;             // [D,E]  bf16
    float* logf  = (float*)(Wdb + (size_t)D * E);              // [M,E]
    float* logi  = logf + (size_t)M * E;                       // [M,E]
    float* logth = logi + (size_t)M * E;                       // [M,E]
    unsigned short* hb = (unsigned short*)(logth + (size_t)M * E); // [M,E] bf16

    // 1) fp32 -> bf16 conversions
    {
        int n4 = (M * D) / 4;
        cvt_f32_bf16_x4<<<(n4 + 255) / 256, 256, 0, stream>>>(
            (const float4*)x, (ushort4*)xb, n4);
        n4 = (3 * E * D) / 4;
        cvt_f32_bf16_x4<<<(n4 + 255) / 256, 256, 0, stream>>>(
            (const float4*)W, (ushort4*)Wb, n4);
        n4 = (D * E) / 4;
        cvt_f32_bf16_x4<<<(n4 + 255) / 256, 256, 0, stream>>>(
            (const float4*)Wd, (ushort4*)Wdb, n4);
    }

    // 2) GEMM1: k = x*W^T + b, fused gate epilogue -> logf/logi/logth
    {
        dim3 grid((3 * E) / BN, M / BM);       // 48 x 128
        wmma_gemm<1><<<grid, 256, 0, stream>>>(xb, Wb, b, logf, logi, logth,
                                               M, 3 * E, D);
    }

    // 3) time scan -> h (bf16)
    {
        int channels = Bb * E;                 // 16384
        scan_kernel<<<(channels + 255) / 256, 256, 0, stream>>>(
            logf, logi, logth, hb, Bb, T, E);
    }

    // 4) GEMM2: out = h * Wd^T
    {
        dim3 grid(D / BN, M / BM);             // 8 x 128
        wmma_gemm<0><<<grid, 256, 0, stream>>>(hb, Wdb, nullptr, out,
                                               nullptr, nullptr, M, D, E);
    }
}
